// NLIEncoder_24335284699634
// MI455X (gfx1250) — matrix operand order repaired and checked by reading, not compiled, not run on hardware
//
#include <hip/hip_runtime.h>
#include <hip/hip_bf16.h>
#include <stdint.h>

typedef __attribute__((ext_vector_type(16))) _Float16 v16h;
typedef __attribute__((ext_vector_type(8)))  _Float16 v8h;
typedef __attribute__((ext_vector_type(8)))  float    v8f;

#define B_    256
#define S_    256
#define D_IN  300
#define D_PAD 320   // 10 * 32
#define H_    200
#define H_PAD 224   // 7 * 32
#define G4    800   // 4*H
#define R_    512   // premise rows 0..255, hypothesis rows 256..511

// ---------------------------------------------------------------------------
// WMMA helpers (CDNA5 16x16x32 f16 -> f32, layouts per ISA 7.12.2)
// ---------------------------------------------------------------------------
__device__ __forceinline__ v8f wmma16(v16h a, v16h b, v8f c) {
  // (neg_a, A, neg_b, B, c_mod, C, reuse_a, reuse_b)
  return __builtin_amdgcn_wmma_f32_16x16x32_f16(false, a, false, b, (short)0, c,
                                                false, false);
}

// A fragment: row-major M x K matrix, tile at (m0, k0), leading dim ld (halfs).
// lane L: M = m0 + (L&15); halves j<8 -> K = khi+j, j>=8 -> K = khi+16+j-8,
// khi = (L&16) ? 8 : 0.
__device__ __forceinline__ v16h load_a_rm(const _Float16* base, int ld, int m0,
                                          int k0) {
  int lane = threadIdx.x & 31;
  const _Float16* p =
      base + (size_t)(m0 + (lane & 15)) * ld + k0 + ((lane & 16) ? 8 : 0);
  v8h lo = *(const v8h*)p;
  v8h hi = *(const v8h*)(p + 16);
  v16h a;
#pragma unroll
  for (int j = 0; j < 8; ++j) { a[j] = lo[j]; a[j + 8] = hi[j]; }
  return a;
}

// B fragment for C = X * W^T: B[k][n] = W[n][k], W row-major [N x ld].
// lane L: N = n0 + (L&15); half j<8 -> K = kb + j, j>=8 -> K = kb+16+j-8,
// kb = (L&16) ? 8 : 0.
__device__ __forceinline__ v16h load_b_t(const _Float16* W, int ld, int n0,
                                         int k0) {
  int lane = threadIdx.x & 31;
  const _Float16* p = W + (size_t)(n0 + (lane & 15)) * ld + k0 +
                      ((lane & 16) ? 8 : 0);
  v8h lo = *(const v8h*)p;
  v8h hi = *(const v8h*)(p + 16);
  v16h b;
#pragma unroll
  for (int j = 0; j < 8; ++j) { b[j] = lo[j]; b[j + 8] = hi[j]; }
  return b;
}

// B fragment for C = A * M: B[k][n] = M[k][n], M row-major [K x ld].
__device__ __forceinline__ v16h load_b_rm(const _Float16* M, int ld, int k0,
                                          int n0) {
  int lane = threadIdx.x & 31;
  int n = n0 + (lane & 15);
  int kb = (lane & 16) ? 8 : 0;
  v16h b;
#pragma unroll
  for (int j = 0; j < 16; ++j) b[j] = M[(size_t)(k0 + kb + j + (j >= 8 ? 8 : 0)) * ld + n];
  return b;
}

// ---------------------------------------------------------------------------
// Prep: f32 weights -> padded f16, fused bias
// ---------------------------------------------------------------------------
__global__ void prep_weights(const float* __restrict__ W_ih,
                             const float* __restrict__ W_hh,
                             const float* __restrict__ b_ih,
                             const float* __restrict__ b_hh,
                             _Float16* __restrict__ Wih16,
                             _Float16* __restrict__ Whh16,
                             float* __restrict__ bias) {
  int idx = blockIdx.x * blockDim.x + threadIdx.x;
  if (idx < G4 * D_PAD) {
    int n = idx / D_PAD, k = idx % D_PAD;
    Wih16[idx] = (k < D_IN) ? (_Float16)W_ih[n * D_IN + k] : (_Float16)0.f;
  }
  if (idx < G4 * H_PAD) {
    int n = idx / H_PAD, k = idx % H_PAD;
    Whh16[idx] = (k < H_) ? (_Float16)W_hh[n * H_ + k] : (_Float16)0.f;
  }
  if (idx < G4) bias[idx] = b_ih[idx] + b_hh[idx];
}

__global__ void zero_words(uint32_t* __restrict__ p, long n) {
  long i = (long)blockIdx.x * blockDim.x + threadIdx.x;
  if (i < n) p[i] = 0u;
}

// Gather embeddings, time-major: Xemb[t][row][k], row<256 premise, else hyp.
__global__ void gather_embed(const int* __restrict__ premise,
                             const int* __restrict__ hypothesis,
                             const float* __restrict__ emb,
                             _Float16* __restrict__ Xemb) {
  int blk = blockIdx.x;           // t*512 + row
  int t = blk >> 9;
  int row = blk & 511;
  int tok = (row < B_) ? premise[row * S_ + t] : hypothesis[(row - B_) * S_ + t];
  int k = threadIdx.x;            // 0..319
  _Float16 v = (k < D_IN) ? (_Float16)emb[(size_t)tok * D_IN + k] : (_Float16)0.f;
  Xemb[((size_t)t * R_ + row) * D_PAD + k] = v;
}

// ---------------------------------------------------------------------------
// LSTM step, part 1: gates = x_t @ Wih^T + h @ Whh^T + (b_ih+b_hh)   [512,800]
// One wave per 16x16 tile; 32 x 50 tiles; 8 waves/block -> 200 blocks.
// ---------------------------------------------------------------------------
__global__ __launch_bounds__(256) void lstm_gates(
    const _Float16* __restrict__ Xt,    // [R_, D_PAD] slab for step t
    const _Float16* __restrict__ Hcur,  // [R_, H_PAD]
    const _Float16* __restrict__ Wih16, // [G4, D_PAD]
    const _Float16* __restrict__ Whh16, // [G4, H_PAD]
    const float* __restrict__ bias,     // [G4]
    float* __restrict__ gates)          // [R_, G4]
{
  int wave = threadIdx.x >> 5;
  int lane = threadIdx.x & 31;
  int tile = blockIdx.x * 8 + wave;     // 0..1599
  int m0 = (tile / 50) * 16;
  int n0 = (tile % 50) * 16;

  float bn = bias[n0 + (lane & 15)];
  v8f c;
#pragma unroll
  for (int r = 0; r < 8; ++r) c[r] = bn;

#pragma unroll
  for (int kx = 0; kx < D_PAD / 32; ++kx)
    c = wmma16(load_a_rm(Xt, D_PAD, m0, kx * 32),
               load_b_t(Wih16, D_PAD, n0, kx * 32), c);
#pragma unroll
  for (int kh = 0; kh < H_PAD / 32; ++kh)
    c = wmma16(load_a_rm(Hcur, H_PAD, m0, kh * 32),
               load_b_t(Whh16, H_PAD, n0, kh * 32), c);

  int rbase = m0 + ((lane & 16) ? 8 : 0);
  int ncol = n0 + (lane & 15);
#pragma unroll
  for (int r = 0; r < 8; ++r)
    gates[(size_t)(rbase + r) * G4 + ncol] = c[r];
}

// LSTM step, part 2: elementwise cell update; writes h (f16) for next step
// and into the hidden-state archive Hall[row][t][:].
__global__ void lstm_cell(const float* __restrict__ gates,
                          float* __restrict__ Cst,      // [R_, H_]
                          _Float16* __restrict__ Hcur,  // [R_, H_PAD]
                          _Float16* __restrict__ Hall,  // [R_, S_, H_PAD]
                          int t) {
  int idx = blockIdx.x * blockDim.x + threadIdx.x;
  if (idx >= R_ * H_) return;
  int row = idx / H_, j = idx % H_;
  const float* g = gates + (size_t)row * G4;
  float gi = g[j], gf = g[H_ + j], gg = g[2 * H_ + j], go = g[3 * H_ + j];
  float si = 1.f / (1.f + expf(-gi));
  float sf = 1.f / (1.f + expf(-gf));
  float so = 1.f / (1.f + expf(-go));
  float c = sf * Cst[idx] + si * tanhf(gg);
  Cst[idx] = c;
  _Float16 h16 = (_Float16)(so * tanhf(c));
  Hcur[(size_t)row * H_PAD + j] = h16;
  Hall[((size_t)row * S_ + t) * H_PAD + j] = h16;
}

// ---------------------------------------------------------------------------
// Fused attention + pooling + classifier. One 512-thread block per batch.
// attn [256x256] f32 lives entirely in LDS (256KB of the 320KB WGP LDS).
// ---------------------------------------------------------------------------
__global__ __launch_bounds__(512) void attention_kernel(
    const _Float16* __restrict__ Hall,
    const float* __restrict__ W_last,   // [3, 800]
    const float* __restrict__ b_last,   // [3]
    float* __restrict__ out)            // [256, 3]
{
  extern __shared__ float smem[];
  float* attnF = smem;                       // 65536 f32
  uint32_t* attnU = (uint32_t*)smem;         // packed {a1 lo16, a2 hi16}
  float* rmax = smem + S_ * S_;              // 256
  float* rinv = rmax + S_;                   // 256
  float* cmax = rinv + S_;                   // 256
  float* cinv = cmax + S_;                   // 256
  float* hqb = cinv + S_;                    // 224
  float* hsb = hqb + H_PAD;                  // 224
  float* lg = hsb + H_PAD;                   // 3

  int b = blockIdx.x;
  const _Float16* Hp = Hall + (size_t)b * S_ * H_PAD;
  const _Float16* Ht = Hall + (size_t)(B_ + b) * S_ * H_PAD;
  int tid = threadIdx.x;
  int wave = tid >> 5, lane = tid & 31;

  // ---- Stage 1: attn = Hp @ Ht^T (M=N=256, K=224), 16 tiles per wave ----
  for (int ti = 0; ti < 16; ++ti) {
    int tile = wave * 16 + ti;
    int m0 = (tile >> 4) * 16, n0 = (tile & 15) * 16;
    v8f c = {};
#pragma unroll
    for (int kk = 0; kk < H_PAD / 32; ++kk)
      c = wmma16(load_a_rm(Hp, H_PAD, m0, kk * 32),
                 load_b_t(Ht, H_PAD, n0, kk * 32), c);
    int rbase = m0 + ((lane & 16) ? 8 : 0);
    int ncol = n0 + (lane & 15);
#pragma unroll
    for (int r = 0; r < 8; ++r) attnF[(rbase + r) * S_ + ncol] = c[r];
  }
  __syncthreads();

  // ---- Stage 2: row stats (threads 0..255) / col stats (256..511) ----
  if (tid < 256) {
    int i = tid;
    float mx = -1e30f;
    for (int j = 0; j < S_; ++j) mx = fmaxf(mx, attnF[i * S_ + j]);
    float s = 0.f;
    for (int j = 0; j < S_; ++j) s += expf(attnF[i * S_ + j] - mx);
    rmax[i] = mx; rinv[i] = 1.f / s;
  } else {
    int j = tid - 256;
    float mx = -1e30f;
    for (int i = 0; i < S_; ++i) mx = fmaxf(mx, attnF[i * S_ + j]);
    float s = 0.f;
    for (int i = 0; i < S_; ++i) s += expf(attnF[i * S_ + j] - mx);
    cmax[j] = mx; cinv[j] = 1.f / s;
  }
  __syncthreads();

  // ---- Stage 2b: pack a1 (low half) and a2 (high half) in place ----
  for (int e = tid; e < S_ * S_; e += 512) {
    int i = e >> 8, j = e & 255;
    float x = attnF[e];
    _Float16 lo = (_Float16)(expf(x - rmax[i]) * rinv[i]);
    _Float16 hi = (_Float16)(expf(x - cmax[j]) * cinv[j]);
    attnU[e] = (uint32_t)__builtin_bit_cast(uint16_t, lo) |
               ((uint32_t)__builtin_bit_cast(uint16_t, hi) << 16);
  }
  __syncthreads();

  // ---- Stage 3: jobs 0..13 -> a1 @ Ht (col-max -> hq),
  //              jobs 14..27 -> a2 @ Hp (col-max -> hs).  K = 256. ----
  for (int job = wave; job < 28; job += 16) {
    bool g1 = job < 14;
    int n0 = (g1 ? job : job - 14) * 16;
    const _Float16* Bsrc = g1 ? Ht : Hp;
    float vmax[8];
#pragma unroll
    for (int r = 0; r < 8; ++r) vmax[r] = -1e30f;

    for (int mt = 0; mt < 16; ++mt) {
      int m0 = mt * 16;
      v8f c = {};
      for (int kk = 0; kk < S_ / 32; ++kk) {
        int k0 = kk * 32;
        int m = m0 + (lane & 15);
        int khi = (lane & 16) ? 8 : 0;
        v16h a;
#pragma unroll
        for (int j = 0; j < 8; ++j) {
          uint32_t w0, w1;
          if (g1) {  // A[m][k] = a1[m][k] (low halves, contiguous in k)
            w0 = attnU[m * S_ + k0 + khi + j];
            w1 = attnU[m * S_ + k0 + khi + 16 + j];
            a[j] = __builtin_bit_cast(_Float16, (uint16_t)(w0 & 0xffffu));
            a[j + 8] = __builtin_bit_cast(_Float16, (uint16_t)(w1 & 0xffffu));
          } else {   // A[m][k] = a2[m][k] = hi half of attn[k][m] (transposed)
            w0 = attnU[(k0 + khi + j) * S_ + m];
            w1 = attnU[(k0 + khi + 16 + j) * S_ + m];
            a[j] = __builtin_bit_cast(_Float16, (uint16_t)(w0 >> 16));
            a[j + 8] = __builtin_bit_cast(_Float16, (uint16_t)(w1 >> 16));
          }
        }
        v16h bb = load_b_rm(Bsrc, H_PAD, k0, n0);
        c = wmma16(a, bb, c);
      }
#pragma unroll
      for (int r = 0; r < 8; ++r) vmax[r] = fmaxf(vmax[r], c[r]);
    }
    float vm = vmax[0];
#pragma unroll
    for (int r = 1; r < 8; ++r) vm = fmaxf(vm, vmax[r]);
    vm = fmaxf(vm, __shfl_xor(vm, 16));   // merge rows 0-7 / 8-15 halves
    if (lane < 16) (g1 ? hqb : hsb)[n0 + lane] = vm;
  }
  __syncthreads();

  // ---- Stage 4: feat = [hq, hs, hq-hs, hq*hs]; logits; log_softmax ----
  if (tid < 3) {
    const float* wl = W_last + tid * G4;
    float acc = b_last[tid];
    for (int h = 0; h < H_; ++h) {
      float q = hqb[h], s = hsb[h];
      acc += wl[h] * q + wl[H_ + h] * s + wl[2 * H_ + h] * (q - s) +
             wl[3 * H_ + h] * (q * s);
    }
    lg[tid] = acc;
  }
  __syncthreads();
  if (tid == 0) {
    float m = fmaxf(lg[0], fmaxf(lg[1], lg[2]));
    float lse = m + logf(expf(lg[0] - m) + expf(lg[1] - m) + expf(lg[2] - m));
    out[b * 3 + 0] = lg[0] - lse;
    out[b * 3 + 1] = lg[1] - lse;
    out[b * 3 + 2] = lg[2] - lse;
  }
}

// ---------------------------------------------------------------------------
extern "C" void kernel_launch(void* const* d_in, const int* in_sizes, int n_in,
                              void* d_out, int out_size, void* d_ws,
                              size_t ws_size, hipStream_t stream) {
  (void)in_sizes; (void)n_in; (void)out_size; (void)ws_size;
  const int* premise = (const int*)d_in[0];
  const int* hypothesis = (const int*)d_in[1];
  const float* emb = (const float*)d_in[2];
  const float* W_ih = (const float*)d_in[3];
  const float* W_hh = (const float*)d_in[4];
  const float* b_ih = (const float*)d_in[5];
  const float* b_hh = (const float*)d_in[6];
  const float* W_last = (const float*)d_in[7];
  const float* b_last = (const float*)d_in[8];
  float* out = (float*)d_out;

  char* ws = (char*)d_ws;
  size_t off = 0;
  auto alloc = [&](size_t bytes) -> void* {
    void* p = ws + off;
    off = (off + bytes + 255) & ~(size_t)255;
    return p;
  };
  _Float16* Wih16 = (_Float16*)alloc((size_t)G4 * D_PAD * 2);
  _Float16* Whh16 = (_Float16*)alloc((size_t)G4 * H_PAD * 2);
  float* bias = (float*)alloc((size_t)G4 * 4);
  _Float16* Hcur = (_Float16*)alloc((size_t)R_ * H_PAD * 2);
  float* Cst = (float*)alloc((size_t)R_ * H_ * 4);
  float* gates = (float*)alloc((size_t)R_ * G4 * 4);
  _Float16* Xemb = (_Float16*)alloc((size_t)S_ * R_ * D_PAD * 2);
  _Float16* Hall = (_Float16*)alloc((size_t)R_ * S_ * H_PAD * 2);

  // Zero-init state + padded archive (padding columns must stay zero).
  {
    long n1 = (long)R_ * H_PAD * 2 / 4;   // Hcur
    long n2 = (long)R_ * H_;              // Cst
    long n3 = (long)R_ * S_ * H_PAD * 2 / 4;  // Hall
    zero_words<<<(unsigned)((n1 + 255) / 256), 256, 0, stream>>>((uint32_t*)Hcur, n1);
    zero_words<<<(unsigned)((n2 + 255) / 256), 256, 0, stream>>>((uint32_t*)Cst, n2);
    zero_words<<<(unsigned)((n3 + 255) / 256), 256, 0, stream>>>((uint32_t*)Hall, n3);
  }

  prep_weights<<<(G4 * D_PAD + 255) / 256, 256, 0, stream>>>(
      W_ih, W_hh, b_ih, b_hh, Wih16, Whh16, bias);

  gather_embed<<<S_ * R_, D_PAD, 0, stream>>>(premise, hypothesis, emb, Xemb);

  // Sequential LSTM: both sequences stacked as 512 rows.
  for (int t = 0; t < S_; ++t) {
    lstm_gates<<<200, 256, 0, stream>>>(Xemb + (size_t)t * R_ * D_PAD, Hcur,
                                        Wih16, Whh16, bias, gates);
    lstm_cell<<<(R_ * H_ + 255) / 256, 256, 0, stream>>>(gates, Cst, Hcur,
                                                         Hall, t);
  }

  // Fused attention/pool/classify: 268KB dynamic LDS per workgroup.
  size_t smemBytes = ((size_t)S_ * S_ + 4 * S_ + 2 * H_PAD + 8) * sizeof(float);
  hipFuncSetAttribute(reinterpret_cast<const void*>(attention_kernel),
                      hipFuncAttributeMaxDynamicSharedMemorySize,
                      (int)smemBytes);
  attention_kernel<<<B_, 512, smemBytes, stream>>>(Hall, W_last, b_last, out);
}
